// Linear4bit_51007031607878
// MI455X (gfx1250) — compile-verified, matrix operand order checked by reading, not run on hardware
//
#include <hip/hip_runtime.h>
#include <stdint.h>

// ---- CDNA5 WMMA vector types ----
typedef __attribute__((ext_vector_type(16))) __bf16 v16bf;
typedef __attribute__((ext_vector_type(2)))  __bf16 v2bf;
typedef __attribute__((ext_vector_type(8)))  float  v8f;

#define IN_F   4096
#define OUT_F  11008
#define M_TOK  2048
#define BM     128
#define BN     128
#define BK     64

// NF4 codebook (QLoRA / bitsandbytes)
__device__ __constant__ float NF4_TAB[16] = {
    -1.0f, -0.6961928009986877f, -0.5250730514526367f, -0.39491748809814453f,
    -0.28444138169288635f, -0.18477343022823334f, -0.09105003625154495f, 0.0f,
    0.07958029955625534f, 0.16093020141124725f, 0.24611230194568634f,
    0.33791524171829224f, 0.44070982933044434f, 0.5626170039176941f,
    0.7229568362236023f, 1.0f};

union FragU { uint32_t u[8]; v16bf v; };

// Packed bf16 multiply (both halves) -- VOP3P V_PK_MUL_BF16, no builtin exposed.
__device__ __forceinline__ uint32_t pk_mul_bf16(uint32_t a, uint32_t b) {
    uint32_t r;
    asm("v_pk_mul_bf16 %0, %1, %2" : "=v"(r) : "v"(a), "v"(b));
    return r;
}

// two f32 -> packed bf16 (low = a, high = b)
__device__ __forceinline__ uint32_t pack_bf16(float a, float b) {
#if __has_builtin(__builtin_amdgcn_cvt_pk_bf16_f32)
    v2bf p = __builtin_amdgcn_cvt_pk_bf16_f32(a, b);
    return __builtin_bit_cast(uint32_t, p);
#else
    return (__float_as_uint(a) >> 16) | (__float_as_uint(b) & 0xFFFF0000u);
#endif
}

__global__ __launch_bounds__(256)
void nf4_gemm_wmma(const float* __restrict__ x,
                   const int*   __restrict__ packed,
                   const float* __restrict__ absmax,
                   const float* __restrict__ bias,
                   float*       __restrict__ y)
{
    // LDS: dequant table + DOUBLE-buffered bf16 tiles (u32 = packed bf16 pair)
    __shared__ __align__(16) uint32_t tab256[256];
    __shared__ __align__(16) uint32_t sA[2 * BM * BK / 2];   // 2 x 16 KB
    __shared__ __align__(16) uint32_t sB[2 * BN * BK / 2];   // 2 x 16 KB

    const int t = threadIdx.x;

    // byte -> packed bf16 pair: low16 = NF4[hi nibble] (element 2p first),
    // high16 = NF4[lo nibble].
    {
        const int hi = t >> 4, lo = t & 15;
        tab256[t] = (__float_as_uint(NF4_TAB[hi]) >> 16) |
                    (__float_as_uint(NF4_TAB[lo]) & 0xFFFF0000u);
    }

    const int Nbase = blockIdx.x * BN;   // out-feature tile
    const int Mbase = blockIdx.y * BM;   // token tile

    const int wave = t >> 5;             // 8 waves: 2 (M) x 4 (N)
    const int lane = t & 31;
    const int wm = wave >> 2;
    const int wn = wave & 3;
    const int l  = lane & 15;
    const int h  = lane >> 4;

    v8f acc[4][2];
#pragma unroll
    for (int f = 0; f < 4; ++f)
#pragma unroll
        for (int g = 0; g < 2; ++g)
#pragma unroll
            for (int r = 0; r < 8; ++r) acc[f][g][r] = 0.0f;

    // cooperative tile-load mapping: each thread owns half a row (32 K elems)
    const int row  = t >> 1;             // 0..127
    const int half = t & 1;              // K 0..31 / 32..63

    const float* xrow = x + (size_t)(Mbase + row) * IN_F + half * 32;
    const int    o    = Nbase + row;
    const int*   prow = packed + (size_t)o * (IN_F / 2) + half * 16;
    const int    ldsoff = row * 32 + half * 16;

    // ---- register staging for the next tile (double buffering) ----
    float4 xr[8]; int4 wr[4]; float sc;

    auto load_regs = [&](int kt) {
        const float4* xp = (const float4*)(xrow + kt);
#pragma unroll
        for (int j = 0; j < 8; ++j) xr[j] = xp[j];
        sc = absmax[o * (IN_F / 64) + (kt >> 6)];
        const int4* pp = (const int4*)(prow + (kt >> 1));
#pragma unroll
        for (int j = 0; j < 4; ++j) wr[j] = pp[j];
    };

    auto store_tile = [&](int buf) {
        uint32_t* sap = sA + buf * 4096 + ldsoff;
        uint32_t* sbp = sB + buf * 4096 + ldsoff;
#pragma unroll
        for (int j = 0; j < 8; ++j) {
            sap[2 * j]     = pack_bf16(xr[j].x, xr[j].y);
            sap[2 * j + 1] = pack_bf16(xr[j].z, xr[j].w);
        }
        const uint32_t sb16 = __float_as_uint(sc) & 0xFFFF0000u;
        const uint32_t spk  = (sb16 >> 16) | sb16;     // bf16 scale, both halves
#pragma unroll
        for (int j4 = 0; j4 < 4; ++j4) {
            const int codes[4] = {wr[j4].x, wr[j4].y, wr[j4].z, wr[j4].w};
#pragma unroll
            for (int c = 0; c < 4; ++c)
                sbp[j4 * 4 + c] = pk_mul_bf16(tab256[codes[c] & 255], spk);
        }
    };

    // prologue: stage tile 0 into buffer 0
    load_regs(0);
    __syncthreads();          // table ready before first dequant reads it
    store_tile(0);
    __syncthreads();

#pragma unroll 1
    for (int kt = 0; kt < IN_F; kt += BK) {
        const int  p        = (kt >> 6) & 1;
        const bool has_next = (kt + BK) < IN_F;

        // (1) issue global loads for tile kt+BK -- overlaps with WMMAs below
        if (has_next) {
            __builtin_prefetch((const void*)(prow + ((kt + 2 * BK) >> 1)), 0, 1);
            load_regs(kt + BK);
        }

        // (2) WMMA on buffer p: two 32-wide K steps, 8 frags each
        const uint32_t* sAc = sA + p * 4096;
        const uint32_t* sBc = sB + p * 4096;
#pragma unroll
        for (int ks = 0; ks < 2; ++ks) {
            v16bf afrag[4], bfrag[2];
#pragma unroll
            for (int f = 0; f < 4; ++f) {
                const uint32_t* pa = sAc + (wm * 64 + f * 16 + l) * 32 + ks * 16 + h * 4;
                FragU u;
                *(uint4*)&u.u[0] = *(const uint4*)(pa);       // K = h*8 .. +7
                *(uint4*)&u.u[4] = *(const uint4*)(pa + 8);   // K = 16+h*8 ..
                afrag[f] = u.v;
            }
#pragma unroll
            for (int g = 0; g < 2; ++g) {
                const uint32_t* pb = sBc + (wn * 32 + g * 16 + l) * 32 + ks * 16 + h * 4;
                FragU u;
                *(uint4*)&u.u[0] = *(const uint4*)(pb);
                *(uint4*)&u.u[4] = *(const uint4*)(pb + 8);
                bfrag[g] = u.v;
            }
#pragma unroll
            for (int f = 0; f < 4; ++f)
#pragma unroll
                for (int g = 0; g < 2; ++g)
                    acc[f][g] = __builtin_amdgcn_wmma_f32_16x16x32_bf16(
                        false, afrag[f], false, bfrag[g],
                        (short)0, acc[f][g], false, false);
        }

        // (3) dequant + store staged tile into the other buffer
        if (has_next) store_tile(p ^ 1);

        // (4) one barrier per iteration: publishes buffer p^1, and guarantees
        //     everyone is done reading buffer p before it is overwritten next
        //     iteration.
        __syncthreads();
    }

    // ----------------------------- epilogue -----------------------------
    // C/D layout: VGPR r holds M = r (lanes 0-15) / r+8 (lanes 16-31),
    // N = lane % 16.
#pragma unroll
    for (int g = 0; g < 2; ++g) {
        const int col = Nbase + wn * 32 + g * 16 + l;
        const float bv = bias[col];
#pragma unroll
        for (int f = 0; f < 4; ++f) {
            const int rbase = Mbase + wm * 64 + f * 16 + 8 * h;
#pragma unroll
            for (int r = 0; r < 8; ++r)
                y[(size_t)(rbase + r) * OUT_F + col] = acc[f][g][r] + bv;
        }
    }
}

extern "C" void kernel_launch(void* const* d_in, const int* in_sizes, int n_in,
                              void* d_out, int out_size, void* d_ws, size_t ws_size,
                              hipStream_t stream) {
    (void)in_sizes; (void)n_in; (void)out_size; (void)d_ws; (void)ws_size;
    const float* x      = (const float*)d_in[0];
    const int*   packed = (const int*)  d_in[1];
    const float* absmax = (const float*)d_in[2];
    const float* bias   = (const float*)d_in[3];
    float*       y      = (float*)d_out;

    dim3 grid(OUT_F / BN, M_TOK / BM);   // 86 x 16
    nf4_gemm_wmma<<<grid, 256, 0, stream>>>(x, packed, absmax, bias, y);
}
